// WaveAttentionWithRoPE_83279415869971
// MI455X (gfx1250) — matrix operand order repaired and checked by reading, not compiled, not run on hardware
//
#include <hip/hip_runtime.h>
#include <math.h>

// ---------------------------------------------------------------------------
// CDNA5 (gfx1250) wave32 WMMA types & helpers
// ---------------------------------------------------------------------------
typedef __attribute__((ext_vector_type(16))) __bf16    v16bf;
typedef __attribute__((ext_vector_type(8)))  float     v8f;
typedef __attribute__((ext_vector_type(4)))  unsigned  u32x4;

__device__ __forceinline__ unsigned short f2bf(float f) {
    union { float f; unsigned u; } x; x.f = f;
    unsigned r = x.u + 0x7FFFu + ((x.u >> 16) & 1u);   // round-to-nearest-even
    return (unsigned short)(r >> 16);
}
__device__ __forceinline__ float bf2f(unsigned short h) {
    union { unsigned u; float f; } x; x.u = ((unsigned)h) << 16; return x.f;
}

// Low 32 bits of the flat address of an LDS object = wave-relative LDS addr
// (flat aperture maps addr[31:0] to LDS).
__device__ __forceinline__ unsigned lds_addr32(const void* p) {
    return (unsigned)(unsigned long long)p;
}
// gfx1250 async DMA: global -> LDS, 16B per lane, tracked by ASYNCcnt.
__device__ __forceinline__ void async_g2l_b128(unsigned lds, const void* g) {
    asm volatile("global_load_async_to_lds_b128 %0, %1, off"
                 :: "v"(lds), "v"((unsigned long long)g) : "memory");
}
__device__ __forceinline__ void wait_async0() {
    asm volatile("s_wait_asynccnt 0x0" ::: "memory");
}

enum { EPI_F32 = 0, EPI_BF16 = 1, EPI_COS_BF16 = 2, EPI_BIAS_F32 = 3, EPI_ATTN_ACC = 4 };

// ---------------------------------------------------------------------------
// LDS-tiled bf16 WMMA GEMM:  C(f32 acc) = A[M,K] * B_T  (+epilogue)
//  - B is stored TRANSPOSED in global memory: [N][K] row-major (ldb = K pitch)
//    so both A and B tiles stage as straight row copies via async DMA.
//  - TRANSA=true: A stored [K,M] (lda = M pitch); staged with a manual
//    transpose (used only for the small kv = k_feat^T v GEMM).
//  - Double-buffered LDS; one barrier + s_wait_asynccnt per K-step.
//  - waves arranged (TBM/32) x (TBN/64); each wave: 2x4 WMMA 16x16 tiles.
//  - batched via blockIdx.z (element strides sA,sB; byte stride sCb).
// All problem dims are multiples of the tile sizes (no edge guards).
// ---------------------------------------------------------------------------
template <int EPI, bool TRANSA, int TBM, int TBN>
__global__ __launch_bounds__((TBM / 32) * (TBN / 64) * 32)
void gemm_bf16(const unsigned short* __restrict__ A,
               const unsigned short* __restrict__ B,
               char* __restrict__ C,
               int Msz, int Nsz, int Ksz, int lda, int ldb, int ldc,
               long sA, long sB, long sCb,
               const float* __restrict__ bias,
               const float* __restrict__ denomB, long sDen,
               const float* __restrict__ lwPtr, float scale, int accFlag) {
    constexpr int WM_ = TBM / 32;
    constexpr int WN_ = TBN / 64;
    constexpr int NTH = WM_ * WN_ * 32;
    constexpr int LDK = 40;   // halfword pitch: 80B rows -> 16B-aligned b128
                              // stores, conflict-free fragment reads

    __shared__ unsigned short sAm[2][TBM * LDK];
    __shared__ unsigned short sBm[2][TBN * LDK];

    const int tid  = threadIdx.x;
    const int lane = tid & 31;
    const int wave = tid >> 5;
    const int wm   = wave % WM_;
    const int wn   = wave / WM_;
    const int half = lane >> 4;     // wave32 lane halves select K/M groups
    const int ml   = lane & 15;

    const int z = blockIdx.z;
    A += (long)z * sA;
    B += (long)z * sB;
    C += (long)z * sCb;
    const float* denom = (EPI == EPI_ATTN_ACC) ? (denomB + (long)z * sDen) : nullptr;

    const long mBlk = (long)blockIdx.y * TBM;
    const long nBlk = (long)blockIdx.x * TBN;

    // ---- stagers (constexpr trip counts -> fully unrolled, no divergence) --
    auto stageA = [&](int k0, int bsel) {
        if constexpr (!TRANSA) {
            constexpr int AIT = (TBM * 4) / NTH;        // 16B vectors
#pragma unroll
            for (int it = 0; it < AIT; it++) {
                int v = tid + it * NTH;
                int row = v >> 2, seg = v & 3;
                async_g2l_b128(lds_addr32(&sAm[bsel][row * LDK + seg * 8]),
                               A + (mBlk + row) * (long)lda + k0 + seg * 8);
            }
        } else {                                        // A stored [K,M]
            constexpr int AIT = (32 * (TBM / 8)) / NTH;
#pragma unroll
            for (int it = 0; it < AIT; it++) {
                int v = tid + it * NTH;
                int krow = v / (TBM / 8), seg = v % (TBM / 8);
                u32x4 w = *(const u32x4*)(A + (long)(k0 + krow) * lda + mBlk + seg * 8);
                union { u32x4 q; unsigned short h[8]; } t; t.q = w;
#pragma unroll
                for (int j = 0; j < 8; j++)
                    sAm[bsel][(seg * 8 + j) * LDK + krow] = t.h[j];
            }
        }
    };
    auto stageB = [&](int k0, int bsel) {
        constexpr int BIT_ = (TBN * 4) / NTH;
#pragma unroll
        for (int it = 0; it < BIT_; it++) {
            int v = tid + it * NTH;
            int row = v >> 2, seg = v & 3;
            async_g2l_b128(lds_addr32(&sBm[bsel][row * LDK + seg * 8]),
                           B + (nBlk + row) * (long)ldb + k0 + seg * 8);
        }
    };

    v8f acc[2][4];
#pragma unroll
    for (int i = 0; i < 2; i++)
#pragma unroll
        for (int j = 0; j < 4; j++)
#pragma unroll
            for (int e = 0; e < 8; e++) acc[i][j][e] = 0.0f;

    const int nk = Ksz / 32;
    stageA(0, 0);
    stageB(0, 0);

    for (int s = 0; s < nk; s++) {
        const int cur = s & 1, nxt = cur ^ 1;
        wait_async0();          // my async loads into `cur` are in LDS
        __syncthreads();        // everyone's are; `nxt` free (reads consumed)
        if (s + 1 < nk) { stageA((s + 1) * 32, nxt); stageB((s + 1) * 32, nxt); }

        // ---- fragments per documented CDNA5 VGPR layouts -------------------
        unsigned af[2][8], bfr[4][8];
#pragma unroll
        for (int sm = 0; sm < 2; sm++) {
            const unsigned short* rp = &sAm[cur][(wm * 32 + sm * 16 + ml) * LDK];
#pragma unroll
            for (int i = 0; i < 8; i++) {
                // A 16x32 bf16: VGPR i -> K = (i/4)*16 + half*8 + (i%4)*2
                int kk = ((i >> 2) << 4) + (half << 3) + ((i & 3) << 1);
                af[sm][i] = *(const unsigned*)(rp + kk);
            }
        }
#pragma unroll
        for (int sn = 0; sn < 4; sn++) {
            const unsigned short* rp = &sBm[cur][(wn * 64 + sn * 16 + ml) * LDK];
#pragma unroll
            for (int i = 0; i < 8; i++) {
                // B 32x16 bf16: VGPR i -> K = (i/4)*16 + half*8 + (i%4)*2 (lane = column)
                int kk = ((i >> 2) << 4) + (half << 3) + ((i & 3) << 1);
                bfr[sn][i] = *(const unsigned*)(rp + kk);
            }
        }
#pragma unroll
        for (int sm = 0; sm < 2; sm++)
#pragma unroll
            for (int sn = 0; sn < 4; sn++) {
                union { unsigned u[8]; v16bf v; } ua, ub;
#pragma unroll
                for (int i = 0; i < 8; i++) { ua.u[i] = af[sm][i]; ub.u[i] = bfr[sn][i]; }
                acc[sm][sn] = __builtin_amdgcn_wmma_f32_16x16x32_bf16(
                    false, ua.v, false, ub.v, (short)0, acc[sm][sn], false, false);
            }
    }

    // ---- epilogue: C layout VGPR r -> m = r + 8*half, n = lane&15 ----------
#pragma unroll
    for (int sm = 0; sm < 2; sm++)
#pragma unroll
        for (int sn = 0; sn < 4; sn++) {
            long n  = nBlk + wn * 64 + sn * 16 + ml;
            long mb = mBlk + wm * 32 + sm * 16 + half * 8;
#pragma unroll
            for (int r = 0; r < 8; r++) {
                long  m = mb + r;
                float v = acc[sm][sn][r];
                if constexpr (EPI == EPI_F32) {
                    ((float*)C)[m * ldc + n] = v;
                } else if constexpr (EPI == EPI_BF16) {
                    ((unsigned short*)C)[m * ldc + n] = f2bf(v);
                } else if constexpr (EPI == EPI_COS_BF16) {
                    ((unsigned short*)C)[m * ldc + n] = f2bf(scale * cosf(v + bias[n]));
                } else if constexpr (EPI == EPI_BIAS_F32) {
                    ((float*)C)[m * ldc + n] = v + bias[n];
                } else {  // EPI_ATTN_ACC: out (+)= lw * acc / max(denom[m],1e-6)
                    float d   = fmaxf(denom[m], 1e-6f);
                    float val = lwPtr[0] * v / d;
                    float* O  = (float*)C;
                    long  o   = m * ldc + n;
                    O[o] = accFlag ? (O[o] + val) : val;
                }
            }
        }
}

// ---------------------------------------------------------------------------
// Elementwise / small kernels
// ---------------------------------------------------------------------------
__global__ void conv_f32_bf16(const float* __restrict__ in,
                              unsigned short* __restrict__ out, long n) {
    long i = (long)blockIdx.x * blockDim.x + threadIdx.x;
    if (i < n) out[i] = f2bf(in[i]);
}

// batched transpose-convert: in [batch][R][C] f32 -> out [batch][C][R] bf16
__global__ void trans_f32_bf16(const float* __restrict__ in,
                               unsigned short* __restrict__ out,
                               int R, int C, long total) {
    long idx = (long)blockIdx.x * blockDim.x + threadIdx.x;
    if (idx >= total) return;
    long rc = (long)R * C;
    long b  = idx / rc;
    long t  = idx % rc;
    int  c  = (int)(t / R);
    int  r  = (int)(t % R);
    out[idx] = f2bf(in[b * rc + (long)r * C + c]);
}

__global__ void softmax3(const float* __restrict__ w, float* __restrict__ lw) {
    if (threadIdx.x == 0) {
        float m  = fmaxf(w[0], fmaxf(w[1], w[2]));
        float e0 = expf(w[0] - m), e1 = expf(w[1] - m), e2 = expf(w[2] - m);
        float s  = e0 + e1 + e2;
        lw[0] = e0 / s; lw[1] = e1 / s; lw[2] = e2 / s;
    }
}

// qkv (bf16, [B*N, 2304]) -> RoPE(q,k) scattered to q/k [B*nH, N, 64] bf16
// and v scattered TRANSPOSED to vT [B*nH, 64, N] bf16 (GEMM B operand).
__global__ void rope_scatter(const unsigned short* __restrict__ qkv,
                             unsigned short* __restrict__ qh,
                             unsigned short* __restrict__ kh,
                             unsigned short* __restrict__ vt,
                             int Np, int nHd, int Wd, long total) {
    long idx = (long)blockIdx.x * blockDim.x + threadIdx.x;
    if (idx >= total) return;
    int  p = (int)(idx & 31);
    long t = idx >> 5;
    int  h = (int)(t % nHd); t /= nHd;
    int  n = (int)(t % Np);
    long b = t / Np;
    int hpos = n / Wd, wpos = n % Wd;
    int fi = p & 15;
    float pos  = (p < 16) ? (float)hpos : (float)wpos;
    float invf = powf(100.0f, -(float)fi * (1.0f / 16.0f));   // ROPE_BASE=100, quarter=16
    float s, c;
    sincosf(pos * invf, &s, &c);
    const unsigned short* rp = qkv + (b * (long)Np + n) * 2304;
    int col = h * 64 + 2 * p;
    float q1 = bf2f(rp[col]),        q2 = bf2f(rp[col + 1]);
    float k1 = bf2f(rp[768 + col]),  k2 = bf2f(rp[768 + col + 1]);
    long ob = ((b * nHd + h) * (long)Np + n) * 64 + 2 * p;
    qh[ob]     = f2bf(q1 * c - q2 * s);
    qh[ob + 1] = f2bf(q1 * s + q2 * c);
    kh[ob]     = f2bf(k1 * c - k2 * s);
    kh[ob + 1] = f2bf(k1 * s + k2 * c);
    long vtb = ((b * nHd + h) * 64 + 2 * p) * (long)Np + n;
    vt[vtb]      = rp[1536 + col];
    vt[vtb + Np] = rp[1536 + col + 1];
}

// Haar 2x2 average, row-major tokens: [BnH, cH, cW, 64] -> [BnH, cH/2, cW/2, 64]
__global__ void haar_down(const unsigned short* __restrict__ in,
                          unsigned short* __restrict__ out,
                          int cH, int cW, long total) {
    long idx = (long)blockIdx.x * blockDim.x + threadIdx.x;
    if (idx >= total) return;
    int  d = (int)(idx & 63);
    long t = idx >> 6;
    int  oW = cW >> 1, oH = cH >> 1;
    int  ox = (int)(t % oW); t /= oW;
    int  oy = (int)(t % oH);
    long bh = t / oH;
    const unsigned short* p = in + ((bh * cH + 2 * oy) * (long)cW + 2 * ox) * 64 + d;
    long rs = (long)cW * 64;
    float s = bf2f(p[0]) + bf2f(p[64]) + bf2f(p[rs]) + bf2f(p[rs + 64]);
    out[idx] = f2bf(0.25f * s);
}

// Haar 2x2 average on transposed layout: [BnH*64, cH, cW] -> [BnH*64, cH/2, cW/2]
__global__ void haar_down_T(const unsigned short* __restrict__ in,
                            unsigned short* __restrict__ out,
                            int cH, int cW, long total) {
    long idx = (long)blockIdx.x * blockDim.x + threadIdx.x;
    if (idx >= total) return;
    int  oW = cW >> 1, oH = cH >> 1;
    long oN = (long)oH * oW;
    long r  = idx / oN;
    long t  = idx % oN;
    int  oy = (int)(t / oW), ox = (int)(t % oW);
    const unsigned short* p = in + r * (long)cH * cW + (2 * oy) * (long)cW + 2 * ox;
    float s = bf2f(p[0]) + bf2f(p[1]) + bf2f(p[cW]) + bf2f(p[cW + 1]);
    out[idx] = f2bf(0.25f * s);
}

// ksum[bh][r] = sum_n k_feat[bh][n][r]   (128 threads/block = one bh)
__global__ void colsum(const unsigned short* __restrict__ kf,
                       float* __restrict__ ksum, int Nj) {
    int bh = blockIdx.x, r = threadIdx.x;
    const unsigned short* p = kf + (long)bh * Nj * 128 + r;
    float s = 0.0f;
    for (int n = 0; n < Nj; n++) s += bf2f(p[(long)n * 128]);
    ksum[bh * 128 + r] = s;
}

// denom[row] = dot(q_feat[row][:], ksum[bh][:])   one 128-thread block per row
__global__ void denom_kernel(const unsigned short* __restrict__ qf,
                             const float* __restrict__ ksum,
                             float* __restrict__ den, int Np) {
    long row = blockIdx.x;
    int  t   = threadIdx.x;
    int  bh  = (int)(row / Np);
    __shared__ float red[128];
    red[t] = bf2f(qf[row * 128 + t]) * ksum[bh * 128 + t];
    __syncthreads();
    for (int s = 64; s > 0; s >>= 1) {
        if (t < s) red[t] += red[t + s];
        __syncthreads();
    }
    if (t == 0) den[row] = red[0];
}

// out_acc f32 [b][h][n][d] -> attn bf16 [(b*N+n)][h*64+d]
__global__ void heads_merge(const float* __restrict__ oacc,
                            unsigned short* __restrict__ attnb,
                            int Np, int nHd, long total) {
    long idx = (long)blockIdx.x * blockDim.x + threadIdx.x;
    if (idx >= total) return;
    int  d = (int)(idx & 63);
    long t = idx >> 6;
    int  n = (int)(t % Np); t /= Np;
    int  h = (int)(t % nHd);
    long b = t / nHd;
    attnb[(b * Np + n) * (long)(nHd * 64) + h * 64 + d] = f2bf(oacc[idx]);
}

// ---------------------------------------------------------------------------
// Host-side launch
// ---------------------------------------------------------------------------
extern "C" void kernel_launch(void* const* d_in, const int* in_sizes, int n_in,
                              void* d_out, int out_size, void* d_ws, size_t ws_size,
                              hipStream_t stream) {
    (void)in_sizes; (void)n_in; (void)out_size; (void)ws_size;
    const float* x        = (const float*)d_in[0];
    const float* Wqkv     = (const float*)d_in[1];
    const float* Wproj    = (const float*)d_in[2];
    const float* bproj    = (const float*)d_in[3];
    const float* omega    = (const float*)d_in[4];
    const float* rff_bias = (const float*)d_in[5];
    const float* level_w  = (const float*)d_in[6];
    float*       out      = (float*)d_out;

    constexpr int  Bb = 16, Hh = 64, Ww = 64, Dd = 768, NHd = 12, HD = 64, Rr = 128;
    constexpr int  Nn = Hh * Ww;             // 4096
    constexpr long Mr = (long)Bb * Nn;       // 65536 token rows
    constexpr int  BH = Bb * NHd;            // 192

    char*  base = (char*)d_ws;
    size_t off  = 0;
    auto alloc = [&](size_t bytes) -> char* {
        char* p = base + off;
        off = (off + bytes + 255) & ~(size_t)255;
        return p;
    };
    float*          lw      = (float*)alloc(16);
    unsigned short* xb      = (unsigned short*)alloc((size_t)Mr * Dd * 2);
    unsigned short* wqkvT   = (unsigned short*)alloc((size_t)Dd * 3 * Dd * 2);   // [2304][768]
    unsigned short* wprojT  = (unsigned short*)alloc((size_t)Dd * Dd * 2);       // [768][768]
    unsigned short* omegaT  = (unsigned short*)alloc((size_t)HD * Rr * 2);       // [128][64]
    unsigned short* qkvb    = (unsigned short*)alloc((size_t)Mr * 3 * Dd * 2);
    unsigned short* qh      = (unsigned short*)alloc((size_t)BH * Nn * HD * 2);
    unsigned short* kh      = (unsigned short*)alloc((size_t)BH * Nn * HD * 2);
    unsigned short* vt      = (unsigned short*)alloc((size_t)BH * HD * Nn * 2);  // [BH][64][N]
    unsigned short* k1      = (unsigned short*)alloc((size_t)BH * 1024 * HD * 2);
    unsigned short* v1t     = (unsigned short*)alloc((size_t)BH * HD * 1024 * 2);
    unsigned short* k2      = (unsigned short*)alloc((size_t)BH * 256 * HD * 2);
    unsigned short* v2t     = (unsigned short*)alloc((size_t)BH * HD * 256 * 2);
    unsigned short* qf      = (unsigned short*)alloc((size_t)BH * Nn * Rr * 2);
    unsigned short* kf      = (unsigned short*)alloc((size_t)BH * Nn * Rr * 2);
    float*          kv      = (float*)alloc((size_t)BH * Rr * HD * 4);
    unsigned short* kvbT    = (unsigned short*)alloc((size_t)BH * HD * Rr * 2);  // [BH][64][128]
    float*          ksum    = (float*)alloc((size_t)BH * Rr * 4);
    float*          den     = (float*)alloc((size_t)BH * Nn * 4);
    float*          oacc    = (float*)alloc((size_t)BH * Nn * HD * 4);
    unsigned short* attnb   = (unsigned short*)alloc((size_t)Mr * Dd * 2);

    softmax3<<<1, 1, 0, stream>>>(level_w, lw);

    // operand conversion: A operands plain, B operands transposed to [N][K]
    long nx = Mr * Dd;
    conv_f32_bf16<<<(nx + 255) / 256, 256, 0, stream>>>(x, xb, nx);
    trans_f32_bf16<<<((long)Dd * 3 * Dd + 255) / 256, 256, 0, stream>>>(Wqkv, wqkvT, Dd, 3 * Dd, (long)Dd * 3 * Dd);
    trans_f32_bf16<<<((long)Dd * Dd + 255) / 256, 256, 0, stream>>>(Wproj, wprojT, Dd, Dd, (long)Dd * Dd);
    trans_f32_bf16<<<((long)HD * Rr + 255) / 256, 256, 0, stream>>>(omega, omegaT, HD, Rr, (long)HD * Rr);

    // QKV projection: [65536,768] x W^T[2304][768] -> bf16 qkv
    gemm_bf16<EPI_BF16, false, 128, 128><<<dim3(18, 512, 1), 256, 0, stream>>>(
        xb, wqkvT, (char*)qkvb, (int)Mr, 3 * Dd, Dd, Dd, Dd, 3 * Dd,
        0, 0, 0, nullptr, nullptr, 0, nullptr, 0.0f, 0);

    // RoPE + head scatter (v written transposed)
    long nrope = (long)Bb * Nn * NHd * 32;
    rope_scatter<<<(nrope + 255) / 256, 256, 0, stream>>>(qkvb, qh, kh, vt, Nn, NHd, Ww, nrope);

    // RFF features for q: [BH*4096,64] x omegaT[128][64], epi scale*cos(.+bias)
    gemm_bf16<EPI_COS_BF16, false, 128, 128><<<dim3(1, (BH * (long)Nn) / 128, 1), 256, 0, stream>>>(
        qh, omegaT, (char*)qf, BH * Nn, Rr, HD, HD, HD, Rr,
        0, 0, 0, rff_bias, nullptr, 0, nullptr, 0.125f, 0);   // sqrt(2/128)=0.125

    // Haar pyramid (2 levels): k row-major, v transposed layout
    long t1 = (long)BH * 1024 * HD, t2 = (long)BH * 256 * HD;
    haar_down  <<<(t1 + 255) / 256, 256, 0, stream>>>(kh, k1, 64, 64, t1);
    haar_down_T<<<(t1 + 255) / 256, 256, 0, stream>>>(vt, v1t, 64, 64, t1);
    haar_down  <<<(t2 + 255) / 256, 256, 0, stream>>>(k1, k2, 32, 32, t2);
    haar_down_T<<<(t2 + 255) / 256, 256, 0, stream>>>(v1t, v2t, 32, 32, t2);

    const unsigned short* kls[3] = {kh, k1, k2};
    const unsigned short* vls[3] = {vt, v1t, v2t};
    const int Njs[3] = {4096, 1024, 256};

    for (int j = 0; j < 3; j++) {
        int  Nj = Njs[j];
        long Mk = (long)BH * Nj;
        // k_feat = scale*cos(k_j @ omega + bias)
        gemm_bf16<EPI_COS_BF16, false, 128, 128><<<dim3(1, Mk / 128, 1), 256, 0, stream>>>(
            kls[j], omegaT, (char*)kf, (int)Mk, Rr, HD, HD, HD, Rr,
            0, 0, 0, rff_bias, nullptr, 0, nullptr, 0.125f, 0);
        // ksum[bh][r]
        colsum<<<BH, 128, 0, stream>>>(kf, ksum, Nj);
        // kv[bh] = k_feat^T @ v_j : A=[Nj][128] (TRANSA), B=vT[64][Nj]
        gemm_bf16<EPI_F32, true, 128, 64><<<dim3(1, 1, BH), 128, 0, stream>>>(
            kf, vls[j], (char*)kv, Rr, HD, Nj, Rr, Nj, HD,
            (long)Nj * Rr, (long)HD * Nj, (long)Rr * HD * 4,
            nullptr, nullptr, 0, nullptr, 0.0f, 0);
        // kv f32 [bh][128][64] -> kvT bf16 [bh][64][128]
        trans_f32_bf16<<<(((long)BH * Rr * HD) + 255) / 256, 256, 0, stream>>>(
            kv, kvbT, Rr, HD, (long)BH * Rr * HD);
        // denom[row] = q_feat . ksum
        denom_kernel<<<BH * Nn, 128, 0, stream>>>(qf, ksum, den, Nn);
        // out (+)= lw[j] * (q_feat @ kv) / denom
        gemm_bf16<EPI_ATTN_ACC, false, 128, 64><<<dim3(1, Nn / 128, BH), 128, 0, stream>>>(
            qf, kvbT, (char*)oacc, Nn, HD, Rr, Rr, Rr, HD,
            (long)Nn * Rr, (long)HD * Rr, (long)Nn * HD * 4,
            nullptr, den, Nn, lw + j, 0.0f, (j > 0) ? 1 : 0);
    }

    // [b][h][n][d] -> [b*n][h*64+d] bf16
    long nm = (long)BH * Nn * HD;
    heads_merge<<<(nm + 255) / 256, 256, 0, stream>>>(oacc, attnb, Nn, NHd, nm);

    // Output projection + bias: [65536,768] x WprojT[768][768] -> f32 out
    gemm_bf16<EPI_BIAS_F32, false, 128, 128><<<dim3(6, 512, 1), 256, 0, stream>>>(
        attnb, wprojT, (char*)out, (int)Mr, Dd, Dd, Dd, Dd, Dd,
        0, 0, 0, bproj, nullptr, 0, nullptr, 0.0f, 0);
}
